// OrbitalMessagePassing_22728966930567
// MI455X (gfx1250) — matrix operand order repaired and checked by reading, not compiled, not run on hardware
//
#include <hip/hip_runtime.h>
#include <hip/hip_bf16.h>
#include <cstdint>
#include <cstddef>

typedef __attribute__((ext_vector_type(16))) _Float16 v16h;
typedef __attribute__((ext_vector_type(8)))  float    v8f;

#define NNODES 20000
#define NEDGES 50000
#define HID    64
#define NCHUNK 130          // 128 bilinear K-chunks (i=0..63, half=0..1) + 2 bias chunks
#define TM     2            // edge sub-tiles (of 16) per wave -> 1563 waves (occupancy)

// workspace byte offsets (all 256-aligned)
#define XF16_OFF   0           // x in f16           [20000*64]  2,560,000 B
#define EHID_OFF   2560000     // relu(edge mlp L1)  [50000*64]  6,400,000 B
#define W2P_OFF    8960000     // W2p frags f16      [130*4*32*16] 532,480 B
#define RWP_OFF    9492480     // root_w frags f16   [2*4*32*16]     8,192 B
#define PRE_OFF    9500672     // agg/pre f32        [20000*64]  5,120,000 B
#define STATS_OFF  14620672    // col sums/sumsq     [128]           512 B

// ---------------- prep kernels ----------------

__global__ void k_zero(float* __restrict__ pre, float* __restrict__ stats) {
  int idx = blockIdx.x * blockDim.x + threadIdx.x;
  if (idx < NNODES * HID) pre[idx] = 0.f;
  if (idx < 128) stats[idx] = 0.f;
}

__global__ void k_prep_xf16(const float* __restrict__ x, _Float16* __restrict__ xf) {
  int idx = blockIdx.x * blockDim.x + threadIdx.x;
  if (idx < NNODES * HID) xf[idx] = (_Float16)x[idx];
}

__global__ void k_prep_ehid(const float* __restrict__ ea, const float* __restrict__ w1,
                            const float* __restrict__ b1, _Float16* __restrict__ eh) {
  int idx = blockIdx.x * blockDim.x + threadIdx.x;
  if (idx < NEDGES * HID) {
    int e = idx >> 6, k = idx & 63;
    float v = ea[e] * w1[k] + b1[k];
    eh[idx] = (_Float16)(v > 0.f ? v : 0.f);
  }
}

// Pack W2p (+ folded b2 rows) into WMMA B-fragment-major order.
// B layout (16-bit, 32x16): element j of v16h is K = (j<8 ? j : j+8) + 8*(lane>=16), col N = lane&15.
__global__ void k_prep_w2p(const float* __restrict__ w2, const float* __restrict__ b2,
                           _Float16* __restrict__ w2p) {
  int u = blockIdx.x * blockDim.x + threadIdx.x;
  if (u < NCHUNK * 4 * 32 * 16) {
    int j = u & 15, lane = (u >> 4) & 31, t = (u >> 9) & 3, c = u >> 11;
    int kc = ((j < 8) ? j : (j + 8)) + ((lane >> 4) << 3);  // 0..31 within chunk
    int o  = t * 16 + (lane & 15);            // output column 0..63
    float v;
    if (c < 128) {
      int i  = c >> 1;                        // x_j index (constant within chunk)
      int kk = ((c & 1) << 5) + kc;           // e_hid index 0..63
      v = w2[(size_t)(i * 64 + o) * 64 + kk];
    } else {
      int i = ((c - 128) << 5) + kc;          // bias chunks: A = x_j, B = b2
      v = b2[(size_t)i * 64 + o];
    }
    w2p[u] = (_Float16)v;
  }
}

__global__ void k_prep_rwp(const float* __restrict__ rw, _Float16* __restrict__ rwp) {
  int u = blockIdx.x * blockDim.x + threadIdx.x;
  if (u < 2 * 4 * 32 * 16) {
    int j = u & 15, lane = (u >> 4) & 31, t = (u >> 9) & 3, c = u >> 11;
    int kc = ((j < 8) ? j : (j + 8)) + ((lane >> 4) << 3);
    int o  = t * 16 + (lane & 15);
    int kk = (c << 5) + kc;
    rwp[u] = (_Float16)rw[(size_t)o * 64 + kk];  // B[k][o] = root_w[o][k]  (x @ root_w.T)
  }
}

// ---------------- main edge GEMM + scatter ----------------
// Per wave: TM sub-tiles of 16 edges. acc[s][t] = 16x16 f32 tile (edges x out-cols).

__launch_bounds__(128)
__global__ void k_edge_gemm(const _Float16* __restrict__ eh,
                            const _Float16* __restrict__ xf,
                            const _Float16* __restrict__ w2p,
                            const int* __restrict__ eidx,   // [2*NEDGES]: src then dst
                            float* __restrict__ pre) {
  __shared__ _Float16 sxj[4][TM][64][16];   // [wave][subtile][i][m] x_j scalars, 16 KB
  const int wave = threadIdx.x >> 5;
  const int lane = threadIdx.x & 31;
  const int m  = lane & 15;
  const int hi = lane >> 4;
  const int kb = hi * 8;
  const int base = (blockIdx.x * 4 + wave) * (16 * TM);
  const bool active = base < NEDGES;

  v16h ehA[TM][2];   // e_hid A-fragments, K halves 0..31 / 32..63, resident
  if (active) {
    #pragma unroll
    for (int s = 0; s < TM; ++s) {
      int e = base + s * 16 + m; if (e >= NEDGES) e = NEDGES - 1;
      const _Float16* er = eh + (size_t)e * HID;
      int sv = eidx[e]; if (sv < 0) sv = 0; if (sv >= NNODES) sv = NNODES - 1;
      const _Float16* xr = xf + (size_t)sv * HID;
      #pragma unroll
      for (int p = 0; p < 16; ++p) {
        int K = ((p < 8) ? p : (p + 8)) + kb;   // A-matrix 16-bit layout
        ehA[s][0][p] = er[K];
        ehA[s][1][p] = er[32 + K];
      }
      for (int jj = 0; jj < 32; ++jj) {         // stage gathered x_j row into LDS
        int col = hi * 32 + jj;
        sxj[wave][s][col][m] = xr[col];
      }
    }
  }
  __syncthreads();
  if (!active) return;

  v8f acc[TM][4] = {};

  for (int i = 0; i < 64; ++i) {
    #pragma unroll
    for (int h = 0; h < 2; ++h) {
      const int c = i * 2 + h;
      const _Float16* bp = w2p + (size_t)c * 2048 + (size_t)lane * 16;
      const v16h B0  = *(const v16h*)(bp);
      const v16h B1  = *(const v16h*)(bp + 512);
      const v16h B2v = *(const v16h*)(bp + 1024);
      const v16h B3  = *(const v16h*)(bp + 1536);
      const _Float16 xs0 = sxj[wave][0][i][m];   // broadcast scalars
      const _Float16 xs1 = sxj[wave][1][i][m];
      v16h A0, A1;                               // virtual Z fragments for both sub-tiles
      #pragma unroll
      for (int p = 0; p < 16; ++p) {
        A0[p] = xs0 * ehA[0][h][p];
        A1[p] = xs1 * ehA[1][h][p];
      }
      acc[0][0] = __builtin_amdgcn_wmma_f32_16x16x32_f16(false, A0, false, B0,  (short)0, acc[0][0], false, false);
      acc[0][1] = __builtin_amdgcn_wmma_f32_16x16x32_f16(false, A0, false, B1,  (short)0, acc[0][1], false, false);
      acc[0][2] = __builtin_amdgcn_wmma_f32_16x16x32_f16(false, A0, false, B2v, (short)0, acc[0][2], false, false);
      acc[0][3] = __builtin_amdgcn_wmma_f32_16x16x32_f16(false, A0, false, B3,  (short)0, acc[0][3], false, false);
      acc[1][0] = __builtin_amdgcn_wmma_f32_16x16x32_f16(false, A1, false, B0,  (short)0, acc[1][0], false, false);
      acc[1][1] = __builtin_amdgcn_wmma_f32_16x16x32_f16(false, A1, false, B1,  (short)0, acc[1][1], false, false);
      acc[1][2] = __builtin_amdgcn_wmma_f32_16x16x32_f16(false, A1, false, B2v, (short)0, acc[1][2], false, false);
      acc[1][3] = __builtin_amdgcn_wmma_f32_16x16x32_f16(false, A1, false, B3,  (short)0, acc[1][3], false, false);
    }
  }

  // bias chunks: A = x_j itself, B = b2 rows
  #pragma unroll
  for (int cc = 0; cc < 2; ++cc) {
    const int c = 128 + cc;
    const _Float16* bp = w2p + (size_t)c * 2048 + (size_t)lane * 16;
    const v16h B0  = *(const v16h*)(bp);
    const v16h B1  = *(const v16h*)(bp + 512);
    const v16h B2v = *(const v16h*)(bp + 1024);
    const v16h B3  = *(const v16h*)(bp + 1536);
    v16h A0, A1;
    #pragma unroll
    for (int p = 0; p < 16; ++p) {
      int K = ((p < 8) ? p : (p + 8)) + kb;
      A0[p] = sxj[wave][0][cc * 32 + K][m];
      A1[p] = sxj[wave][1][cc * 32 + K][m];
    }
    acc[0][0] = __builtin_amdgcn_wmma_f32_16x16x32_f16(false, A0, false, B0,  (short)0, acc[0][0], false, false);
    acc[0][1] = __builtin_amdgcn_wmma_f32_16x16x32_f16(false, A0, false, B1,  (short)0, acc[0][1], false, false);
    acc[0][2] = __builtin_amdgcn_wmma_f32_16x16x32_f16(false, A0, false, B2v, (short)0, acc[0][2], false, false);
    acc[0][3] = __builtin_amdgcn_wmma_f32_16x16x32_f16(false, A0, false, B3,  (short)0, acc[0][3], false, false);
    acc[1][0] = __builtin_amdgcn_wmma_f32_16x16x32_f16(false, A1, false, B0,  (short)0, acc[1][0], false, false);
    acc[1][1] = __builtin_amdgcn_wmma_f32_16x16x32_f16(false, A1, false, B1,  (short)0, acc[1][1], false, false);
    acc[1][2] = __builtin_amdgcn_wmma_f32_16x16x32_f16(false, A1, false, B2v, (short)0, acc[1][2], false, false);
    acc[1][3] = __builtin_amdgcn_wmma_f32_16x16x32_f16(false, A1, false, B3,  (short)0, acc[1][3], false, false);
  }

  // scatter-add messages into per-node accumulator (C layout: VGPR r -> row r+8*hi, col = lane&15)
  #pragma unroll
  for (int s = 0; s < TM; ++s) {
    #pragma unroll
    for (int r = 0; r < 8; ++r) {
      int e = base + s * 16 + r + hi * 8;
      if (e < NEDGES) {
        int d = eidx[NEDGES + e]; if (d < 0) d = 0; if (d >= NNODES) d = NNODES - 1;
        float* dp = pre + (size_t)d * HID + (lane & 15);
        atomicAdd(dp +  0, acc[s][0][r]);
        atomicAdd(dp + 16, acc[s][1][r]);
        atomicAdd(dp + 32, acc[s][2][r]);
        atomicAdd(dp + 48, acc[s][3][r]);
      }
    }
  }
}

// ---------------- root linear (WMMA) + bias, in-place on pre ----------------

__launch_bounds__(128)
__global__ void k_root_gemm(const _Float16* __restrict__ xf,
                            const _Float16* __restrict__ rwp,
                            const float* __restrict__ cbias,
                            float* __restrict__ pre) {
  const int wave = threadIdx.x >> 5, lane = threadIdx.x & 31;
  const int m = lane & 15, hi = lane >> 4, kb = hi * 8;
  const int tile = blockIdx.x * 4 + wave;
  if (tile * 16 >= NNODES) return;
  const int nb = tile * 16;
  const _Float16* xr = xf + (size_t)(nb + m) * HID;
  v16h A0, A1;
  #pragma unroll
  for (int p = 0; p < 16; ++p) {
    int K = ((p < 8) ? p : (p + 8)) + kb;
    A0[p] = xr[K];
    A1[p] = xr[32 + K];
  }
  v8f acc[4] = {};
  #pragma unroll
  for (int c = 0; c < 2; ++c) {
    #pragma unroll
    for (int t = 0; t < 4; ++t) {
      const v16h B = *(const v16h*)(rwp + ((size_t)(c * 4 + t) * 32 + lane) * 16);
      acc[t] = __builtin_amdgcn_wmma_f32_16x16x32_f16(false, (c ? A1 : A0), false, B, (short)0, acc[t], false, false);
    }
  }
  #pragma unroll
  for (int r = 0; r < 8; ++r) {
    int row = nb + r + hi * 8;
    #pragma unroll
    for (int t = 0; t < 4; ++t) {
      int col = t * 16 + (lane & 15);
      size_t idx = (size_t)row * HID + col;
      pre[idx] = pre[idx] + acc[t][r] + cbias[col];
    }
  }
}

// ---------------- BatchNorm stats + finalize ----------------

__global__ void k_stats(const float* __restrict__ pre, float* __restrict__ stats) {
  __shared__ float ss[256], sq[256];
  const int col = threadIdx.x & 63;
  const int rg  = threadIdx.x >> 6;   // 0..3 row groups per block
  float s = 0.f, q = 0.f;
  for (int row = blockIdx.x * 4 + rg; row < NNODES; row += gridDim.x * 4) {
    float v = pre[(size_t)row * HID + col];
    s += v; q += v * v;
  }
  ss[threadIdx.x] = s; sq[threadIdx.x] = q;
  __syncthreads();
  if (rg == 0) {
    s = ss[col] + ss[col + 64] + ss[col + 128] + ss[col + 192];
    q = sq[col] + sq[col + 64] + sq[col + 128] + sq[col + 192];
    atomicAdd(&stats[col], s);
    atomicAdd(&stats[64 + col], q);
  }
}

__global__ void k_final(const float* __restrict__ pre, const float* __restrict__ x,
                        const float* __restrict__ stats,
                        const float* __restrict__ gamma, const float* __restrict__ beta,
                        float* __restrict__ out) {
  int idx = blockIdx.x * blockDim.x + threadIdx.x;
  if (idx >= NNODES * HID) return;
  int col = idx & 63;
  float mean = stats[col] * (1.f / NNODES);
  float var  = stats[64 + col] * (1.f / NNODES) - mean * mean;
  float y = (pre[idx] - mean) * rsqrtf(var + 1e-5f) * gamma[col] + beta[col];
  out[idx] = x[idx] + (y > 0.f ? y : 0.f);
}

// ---------------- launcher ----------------

extern "C" void kernel_launch(void* const* d_in, const int* in_sizes, int n_in,
                              void* d_out, int out_size, void* d_ws, size_t ws_size,
                              hipStream_t stream) {
  const float* x   = (const float*)d_in[0];
  const float* ea  = (const float*)d_in[1];
  const int*   ei  = (const int*)d_in[2];
  const float* w1  = (const float*)d_in[3];
  const float* b1  = (const float*)d_in[4];
  const float* w2  = (const float*)d_in[5];
  const float* b2  = (const float*)d_in[6];
  const float* rw  = (const float*)d_in[7];
  const float* cb  = (const float*)d_in[8];
  const float* g   = (const float*)d_in[9];
  const float* bt  = (const float*)d_in[10];
  float* out = (float*)d_out;

  char* ws = (char*)d_ws;
  _Float16* xf   = (_Float16*)(ws + XF16_OFF);
  _Float16* eh   = (_Float16*)(ws + EHID_OFF);
  _Float16* w2p  = (_Float16*)(ws + W2P_OFF);
  _Float16* rwp  = (_Float16*)(ws + RWP_OFF);
  float*    pre  = (float*)(ws + PRE_OFF);
  float*    stat = (float*)(ws + STATS_OFF);

  k_zero<<<(NNODES * HID + 255) / 256, 256, 0, stream>>>(pre, stat);
  k_prep_xf16<<<(NNODES * HID + 255) / 256, 256, 0, stream>>>(x, xf);
  k_prep_ehid<<<(NEDGES * HID + 255) / 256, 256, 0, stream>>>(ea, w1, b1, eh);
  k_prep_w2p<<<(NCHUNK * 2048 + 255) / 256, 256, 0, stream>>>(w2, b2, w2p);
  k_prep_rwp<<<(2 * 2048 + 255) / 256, 256, 0, stream>>>(rw, rwp);

  const int nwaves = (NEDGES + 16 * TM - 1) / (16 * TM);   // 1563
  k_edge_gemm<<<(nwaves + 3) / 4, 128, 0, stream>>>(eh, xf, w2p, ei, pre);

  const int ntiles = NNODES / 16;                          // 1250
  k_root_gemm<<<(ntiles + 3) / 4, 128, 0, stream>>>(xf, rwp, cb, pre);

  k_stats<<<128, 256, 0, stream>>>(pre, stat);
  k_final<<<(NNODES * HID + 255) / 256, 256, 0, stream>>>(pre, x, stat, g, bt, out);
}
